// AttentionalDynamicsUpdate_88175678587133
// MI455X (gfx1250) — matrix operand order repaired and checked by reading, not compiled, not run on hardware
//
#include <hip/hip_runtime.h>

// ---------------------------------------------------------------------------
// MI455X (gfx1250, wave32) fused attention block.
// All GEMMs run on v_wmma_f32_16x16x32_bf16. Weights are pre-swizzled into
// WMMA B-fragment tile order by prep_weights (they then stay hot in the
// 192MB L2 and are re-read by all 256 blocks). Activations staged in LDS.
// Projection loops are register-blocked (2x8 / 2x2 output tiles per wave)
// so each A/B fragment load feeds multiple WMMAs.
// ---------------------------------------------------------------------------

typedef __attribute__((ext_vector_type(16))) __bf16 v16bf;
typedef __attribute__((ext_vector_type(8)))  float  v8f;
typedef __attribute__((ext_vector_type(4)))  int    i4;

union Frag16 { v16bf v; i4 q[2]; };
union Acc8   { v8f  v; float f[8]; };

__device__ __forceinline__ unsigned pack_bf16x2(float a, float b) {
    unsigned ua = __float_as_uint(a), ub = __float_as_uint(b);
    ua = (ua + 0x7FFFu + ((ua >> 16) & 1u)) >> 16;
    ub = (ub + 0x7FFFu + ((ub >> 16) & 1u)) >> 16;
    return (ua & 0xFFFFu) | (ub << 16);
}
__device__ __forceinline__ unsigned short bf16_1(float a) {
    unsigned ua = __float_as_uint(a);
    return (unsigned short)((ua + 0x7FFFu + ((ua >> 16) & 1u)) >> 16);
}

__device__ __forceinline__ v8f wmma_bf16(const Frag16& a, const Frag16& b, v8f c) {
    return __builtin_amdgcn_wmma_f32_16x16x32_bf16(false, a.v, false, b.v,
                                                   (short)0, c, false, false);
}

// A fragment (16x32 bf16) from an LDS row-major matrix, row stride rs elems.
// lane<16: M=lane, packs K 0..7 / 16..23 ; lane>=16: M=lane-16, K 8..15 / 24..31
__device__ __forceinline__ Frag16 load_a_lds(const char* base, int m0, int k0,
                                             int rs, int lane) {
    Frag16 f;
    const int lrow = lane & 15, kh = lane >> 4;
    const char* p = base + ((size_t)(m0 + lrow) * rs + k0) * 2 + kh * 16;
    f.q[0] = *(const i4*)(p);
    f.q[1] = *(const i4*)(p + 32);
    return f;
}
// B fragment (32x16 bf16) from an LDS row-major [K][N] matrix (lane = K row,
// VGPR v packs N pairs -> 32 contiguous bytes of one row).
__device__ __forceinline__ Frag16 load_b_lds(const char* base, int k0, int n0,
                                             int rs, int lane) {
    Frag16 f;
    const char* p = base + ((size_t)(k0 + lane) * rs + n0) * 2;
    f.q[0] = *(const i4*)(p);
    f.q[1] = *(const i4*)(p + 16);
    return f;
}
// B fragment from pre-swizzled global weight tiles (1KB per 16x32 tile,
// 32B contiguous per lane).
__device__ __forceinline__ Frag16 load_b_glb(const unsigned short* w, int ft,
                                             int kt, int nkt, int lane) {
    Frag16 f;
    const char* p = (const char*)(w + ((size_t)(ft * nkt + kt) << 9) + lane * 16);
    f.q[0] = *(const i4*)(p);
    f.q[1] = *(const i4*)(p + 16);
    return f;
}

// ------------------------- weight pre-swizzle ------------------------------
// ws layout (bf16): [0]=WK (2048x512), [1M]=WQ (2048x512), [2M]=WV (2048x512),
// [3M]=WO (512x2048). Each stored as tiles (ft,kt): 32 lanes x 16 bf16 where
// element s=2v+p = W[f0+lane%16, k0+8*(lane/16)+(s&7)+16*(s>>3)].
__global__ __launch_bounds__(256) void prep_weights(
    const float* __restrict__ W_hk, const float* __restrict__ W_hv,
    const float* __restrict__ W_zk, const float* __restrict__ W_zv,
    const float* __restrict__ W_q,  const float* __restrict__ W_out,
    unsigned short* __restrict__ ws)
{
    int t   = blockIdx.x * 256 + threadIdx.x;      // 0 .. 4M-1
    int buf = t >> 20;
    int r   = t & 0xFFFFF;
    int p    = r & 1;
    int v    = (r >> 1) & 7;
    int lane = (r >> 4) & 31;
    int tile = r >> 9;
    float val;
    if (buf < 3) {                                  // 512-deep matmuls
        int kt = tile & 15, ft = tile >> 4;
        int f = ft * 16 + (lane & 15), k = kt * 32 + (lane >> 4) * 8 + ((2 * v + p) & 7) + ((2 * v + p) >> 3) * 16;
        if (buf == 0)      val = (f < 1024) ? W_hk[f * 512 + k] : W_zk[(f - 1024) * 512 + k];
        else if (buf == 1) val = W_q[f * 512 + k];
        else               val = (f < 1024) ? W_hv[f * 512 + k] : W_zv[(f - 1024) * 512 + k];
    } else {                                        // W_out: 512 x 2048
        int kt = tile & 63, ft = tile >> 6;
        int f = ft * 16 + (lane & 15), k = kt * 32 + (lane >> 4) * 8 + ((2 * v + p) & 7) + ((2 * v + p) >> 3) * 16;
        val = W_out[f * 2048 + k];
    }
    ws[t] = bf16_1(val);
}

// ------------------------------ main kernel --------------------------------
// LDS map (bytes):
//   0      XH  bf16[64][512]   (65536)  \__ overlaid by OUT f32[64][512] in stage E
//   65536  XZ  bf16[64][512]   (65536)  /
//   131072 CA  bf16[64][256]   (32768)  Kc / Vc   \__ overlaid by U^T bf16[512][64]
//   163840 CB  bf16[64][256]   (32768)  Qc       /
//   196608 LOG f32[64][64]     (16384)
//   212992 PB  bf16[64][64]    (8192)
#define LDS_BYTES 221184

__global__ __launch_bounds__(256) void attn_main(
    const float* __restrict__ h, const float* __restrict__ z,
    const unsigned short* __restrict__ WK, const unsigned short* __restrict__ WQ,
    const unsigned short* __restrict__ WV, const unsigned short* __restrict__ WO,
    const float* __restrict__ ln_g, const float* __restrict__ ln_b,
    float* __restrict__ outg)
{
    __shared__ i4 smem4[LDS_BYTES / 16];
    char* smem = (char*)smem4;
    char* XH = smem;
    char* XZ = smem + 65536;
    char* CA = smem + 131072;
    char* CB = smem + 163840;
    char* UB = smem + 131072;
    float* LOG = (float*)(smem + 196608);
    char* PB = smem + 212992;

    const int n    = blockIdx.x;
    const int tid  = threadIdx.x;
    const int wave = tid >> 5, lane = tid & 31;
    const int lcol = lane & 15, rbase = (lane >> 4) * 8;
    const int mhalf = wave >> 2;          // 0/1 -> rows [mhalf*32, +32)
    const int grp   = wave & 3;           // stage-B: 0,1 -> K tiles ; 2,3 -> Q tiles

    // ---- Stage A: h[n], z[n] fp32 -> bf16 in LDS ----
    {
        const float4* h4 = (const float4*)(h + (size_t)n * 32768);
        const float4* z4 = (const float4*)(z + (size_t)n * 32768);
        unsigned* xh = (unsigned*)XH;
        unsigned* xz = (unsigned*)XZ;
        for (int it = 0; it < 32; ++it) {
            int v = tid + it * 256;                 // float4 index, 0..8191
            float4 a = h4[v];
            xh[v * 2]     = pack_bf16x2(a.x, a.y);
            xh[v * 2 + 1] = pack_bf16x2(a.z, a.w);
            float4 b = z4[v];
            xz[v * 2]     = pack_bf16x2(b.x, b.y);
            xz[v * 2 + 1] = pack_bf16x2(b.z, b.w);
        }
    }
    __syncthreads();

    // ---- Stage B: K/Q chunk projections (register-blocked 2x8) + logits ----
    Acc8 lacc[2] = {};
    const bool isQw = grp >= 2;
    const int  ntb  = (grp & 1) * 8;      // n-tile base within the kind
    for (int c = 0; c < 8; ++c) {
        const char* XK = (c < 4) ? XH : XZ;
        const char* Xs = isQw ? XH : XK;
        const unsigned short* Wb = isQw ? WQ : WK;
        Acc8 acc[2][8] = {};
#pragma unroll 4
        for (int kt = 0; kt < 16; ++kt) {
            Frag16 a0 = load_a_lds(Xs, mhalf * 32,      kt * 32, 512, lane);
            Frag16 a1 = load_a_lds(Xs, mhalf * 32 + 16, kt * 32, 512, lane);
#pragma unroll
            for (int j = 0; j < 8; ++j) {
                Frag16 b = load_b_glb(Wb, c * 16 + ntb + j, kt, 16, lane);
                acc[0][j].v = wmma_bf16(a0, b, acc[0][j].v);
                acc[1][j].v = wmma_bf16(a1, b, acc[1][j].v);
            }
        }
        if (!isQw) {        // Kc row-major [64][256]
#pragma unroll
            for (int i = 0; i < 2; ++i)
#pragma unroll
                for (int j = 0; j < 8; ++j) {
                    int mt = mhalf * 2 + i, nt = ntb + j;
#pragma unroll
                    for (int r2 = 0; r2 < 8; ++r2)
                        ((unsigned short*)CA)[(mt * 16 + rbase + r2) * 256 + nt * 16 + lcol] =
                            bf16_1(acc[i][j].f[r2]);
                }
        } else {            // Qc row-major [64 q][256 feat]
#pragma unroll
            for (int i = 0; i < 2; ++i)
#pragma unroll
                for (int j = 0; j < 8; ++j) {
                    int mt = mhalf * 2 + i, nt = ntb + j;
#pragma unroll
                    for (int r2 = 0; r2 < 8; ++r2)
                        ((unsigned short*)CB)[(mt * 16 + rbase + r2) * 256 + nt * 16 + lcol] =
                            bf16_1(acc[i][j].f[r2]);
                }
        }
        __syncthreads();
        // logits += Kc @ QcT  (2 tiles per wave)
#pragma unroll
        for (int li = 0; li < 2; ++li) {
            int lt = wave * 2 + li;
            int mt = lt >> 2, nt = lt & 3;
#pragma unroll
            for (int et = 0; et < 8; ++et) {
                Frag16 a = load_a_lds(CA, mt * 16, et * 32, 256, lane);
                Frag16 b = load_a_lds(CB, nt * 16, et * 32, 256, lane);
                lacc[li].v = wmma_bf16(a, b, lacc[li].v);
            }
        }
        __syncthreads();
    }
#pragma unroll
    for (int li = 0; li < 2; ++li) {
        int lt = wave * 2 + li, mt = lt >> 2, nt = lt & 3;
#pragma unroll
        for (int r2 = 0; r2 < 8; ++r2)
            LOG[(mt * 16 + rbase + r2) * 64 + nt * 16 + lcol] = lacc[li].f[r2];
    }
    __syncthreads();

    // ---- Stage C: softmax over q (rows = key slots), scale 1/sqrt(512) ----
    {
        const float scale = 0.04419417382415922f;
        unsigned short* P = (unsigned short*)PB;
        for (int r2 = 0; r2 < 8; ++r2) {
            int row = wave * 8 + r2;
            float x0 = LOG[row * 64 + lane] * scale;
            float x1 = LOG[row * 64 + 32 + lane] * scale;
            float m = fmaxf(x0, x1);
            for (int off = 16; off; off >>= 1) m = fmaxf(m, __shfl_xor(m, off, 32));
            float e0 = __expf(x0 - m), e1 = __expf(x1 - m);
            float s = e0 + e1;
            for (int off = 16; off; off >>= 1) s += __shfl_xor(s, off, 32);
            float inv = 1.0f / s;
            P[row * 64 + lane]      = bf16_1(e0 * inv);
            P[row * 64 + 32 + lane] = bf16_1(e1 * inv);
        }
    }
    __syncthreads();

    // ---- Stage D: U = v_cat @ W_out^T, each wave owns 64 output columns ----
    Acc8 uacc[16] = {};
    for (int c = 0; c < 8; ++c) {
        const char* Xs = (c < 4) ? XH : XZ;
        // Vc [64][256] -> CA ; 2x2 register blocks, two passes per wave
#pragma unroll
        for (int pass = 0; pass < 2; ++pass) {
            Acc8 acc[2][2] = {};
#pragma unroll 4
            for (int kt = 0; kt < 16; ++kt) {
                Frag16 a0 = load_a_lds(Xs, mhalf * 32,      kt * 32, 512, lane);
                Frag16 a1 = load_a_lds(Xs, mhalf * 32 + 16, kt * 32, 512, lane);
#pragma unroll
                for (int j = 0; j < 2; ++j) {
                    int nt = grp * 4 + pass * 2 + j;
                    Frag16 b = load_b_glb(WV, c * 16 + nt, kt, 16, lane);
                    acc[0][j].v = wmma_bf16(a0, b, acc[0][j].v);
                    acc[1][j].v = wmma_bf16(a1, b, acc[1][j].v);
                }
            }
#pragma unroll
            for (int i = 0; i < 2; ++i)
#pragma unroll
                for (int j = 0; j < 2; ++j) {
                    int mt = mhalf * 2 + i, nt = grp * 4 + pass * 2 + j;
#pragma unroll
                    for (int r2 = 0; r2 < 8; ++r2)
                        ((unsigned short*)CA)[(mt * 16 + rbase + r2) * 256 + nt * 16 + lcol] =
                            bf16_1(acc[i][j].f[r2]);
                }
        }
        __syncthreads();
        // U stripe += Vc @ WO_chunk^T  (B reused across 4 q-tiles)
        for (int et = 0; et < 8; ++et) {
            Frag16 bfr[4];
#pragma unroll
            for (int dt = 0; dt < 4; ++dt)
                bfr[dt] = load_b_glb(WO, wave * 4 + dt, c * 8 + et, 64, lane);
#pragma unroll
            for (int qt = 0; qt < 4; ++qt) {
                Frag16 a = load_a_lds(CA, qt * 16, et * 32, 256, lane);
#pragma unroll
                for (int dt = 0; dt < 4; ++dt)
                    uacc[qt * 4 + dt].v = wmma_bf16(a, bfr[dt], uacc[qt * 4 + dt].v);
            }
        }
        __syncthreads();
    }
    // Spill U^T as bf16 [512][64] (overlays CA+CB)
#pragma unroll
    for (int qt = 0; qt < 4; ++qt)
#pragma unroll
        for (int dt = 0; dt < 4; ++dt)
#pragma unroll
            for (int r2 = 0; r2 < 8; ++r2)
                ((unsigned short*)UB)[(wave * 64 + dt * 16 + lcol) * 64 +
                                      qt * 16 + rbase + r2] =
                    bf16_1(uacc[qt * 4 + dt].f[r2]);
    __syncthreads();

    // ---- Stage E: out = probs @ U ----
    Acc8 oacc[16] = {};
#pragma unroll
    for (int kt2 = 0; kt2 < 2; ++kt2) {
        Frag16 bfr[4];
#pragma unroll
        for (int dt = 0; dt < 4; ++dt)
            bfr[dt] = load_a_lds(UB, wave * 64 + dt * 16, kt2 * 32, 64, lane);
#pragma unroll
        for (int mt = 0; mt < 4; ++mt) {
            Frag16 a = load_a_lds(PB, mt * 16, kt2 * 32, 64, lane);
#pragma unroll
            for (int dt = 0; dt < 4; ++dt)
                oacc[mt * 4 + dt].v = wmma_bf16(a, bfr[dt], oacc[mt * 4 + dt].v);
        }
    }
    // out f32 [64][512] overlays XH/XZ (no outstanding readers of X)
    float* OUT = (float*)smem;
#pragma unroll
    for (int mt = 0; mt < 4; ++mt)
#pragma unroll
        for (int dt = 0; dt < 4; ++dt)
#pragma unroll
            for (int r2 = 0; r2 < 8; ++r2)
                OUT[(mt * 16 + rbase + r2) * 512 + wave * 64 + dt * 16 + lcol] =
                    oacc[mt * 4 + dt].f[r2];
    __syncthreads();

    // ---- Stage F: layernorm (fp32), 8 rows per wave ----
    {
        const float invD = 1.0f / 512.0f;
        float* og = outg + (size_t)n * 32768;
        for (int r2 = 0; r2 < 8; ++r2) {
            int row = wave * 8 + r2;
            float s = 0.f, sq = 0.f, vals[16];
#pragma unroll
            for (int j = 0; j < 16; ++j) {
                float x = OUT[row * 512 + j * 32 + lane];
                vals[j] = x; s += x; sq += x * x;
            }
            for (int off = 16; off; off >>= 1) {
                s  += __shfl_xor(s, off, 32);
                sq += __shfl_xor(sq, off, 32);
            }
            float mu  = s * invD;
            float var = sq * invD - mu * mu;
            float rstd = rsqrtf(var + 1e-5f);
#pragma unroll
            for (int j = 0; j < 16; ++j) {
                int e = j * 32 + lane;
                og[row * 512 + e] = (vals[j] - mu) * rstd * ln_g[e] + ln_b[e];
            }
        }
    }
}

// ------------------------------- launcher ----------------------------------
extern "C" void kernel_launch(void* const* d_in, const int* in_sizes, int n_in,
                              void* d_out, int out_size, void* d_ws, size_t ws_size,
                              hipStream_t stream) {
    (void)in_sizes; (void)n_in; (void)out_size; (void)ws_size; // ws needs >= 8MB
    const float* h     = (const float*)d_in[0];
    const float* z     = (const float*)d_in[1];
    const float* W_hk  = (const float*)d_in[2];
    const float* W_hv  = (const float*)d_in[3];
    const float* W_zk  = (const float*)d_in[4];
    const float* W_zv  = (const float*)d_in[5];
    const float* W_q   = (const float*)d_in[6];
    const float* W_out = (const float*)d_in[7];
    const float* ln_g  = (const float*)d_in[8];
    const float* ln_b  = (const float*)d_in[9];

    unsigned short* ws = (unsigned short*)d_ws;
    // 4M bf16 elements total: WK | WQ | WV | WO, 1M each.
    prep_weights<<<16384, 256, 0, stream>>>(W_hk, W_hv, W_zk, W_zv, W_q, W_out, ws);

    const unsigned short* WK = ws;
    const unsigned short* WQ = ws + (1u << 20);
    const unsigned short* WV = ws + (2u << 20);
    const unsigned short* WO = ws + (3u << 20);
    attn_main<<<256, 256, 0, stream>>>(h, z, WK, WQ, WV, WO, ln_g, ln_b,
                                       (float*)d_out);
}
